// MaxSim_20907900797344
// MI455X (gfx1250) — matrix operand order repaired and checked by reading, not compiled, not run on hardware
//
#include <hip/hip_runtime.h>
#include <hip/hip_bf16.h>

#define BATCH    16
#define QLEN     32
#define DIM      128
#define DOC_LEN  128
#define NTOPK    1024
#define KOUT     10

typedef __attribute__((ext_vector_type(16))) __bf16 v16bf;
typedef __attribute__((ext_vector_type(8)))  float  v8f;
typedef __attribute__((ext_vector_type(4)))  float  f4;

// ---------------------------------------------------------------------------
// Kernel 1: per-row sort of pids + duplicate marking (-1), one block per row.
// Matches reference _unique_pids: sorted pids, later duplicates -> -1.
// ---------------------------------------------------------------------------
__global__ __launch_bounds__(256) void sort_dedup_kernel(
    const int* __restrict__ topk_idx, const int* __restrict__ emb2pid,
    int* __restrict__ pids_out)
{
    __shared__ int s[NTOPK];
    const int b = blockIdx.x;
    const int tid = threadIdx.x;

    for (int i = tid; i < NTOPK; i += 256)
        s[i] = emb2pid[topk_idx[b * NTOPK + i]];
    __syncthreads();

    // bitonic sort, ascending
    for (int kk = 2; kk <= NTOPK; kk <<= 1) {
        for (int jj = kk >> 1; jj >= 1; jj >>= 1) {
            for (int t = tid; t < NTOPK / 2; t += 256) {
                int i = ((t & ~(jj - 1)) << 1) | (t & (jj - 1));
                int p = i | jj;
                bool asc = ((i & kk) == 0);
                int a = s[i], c = s[p];
                if (asc ? (a > c) : (a < c)) { s[i] = c; s[p] = a; }
            }
            __syncthreads();
        }
    }

    for (int i = tid; i < NTOPK; i += 256) {
        int v = s[i];
        pids_out[b * NTOPK + i] = (i > 0 && s[i - 1] == v) ? -1 : v;
    }
}

// ---------------------------------------------------------------------------
// fp32 -> bf16_hi + bf16_lo split (Ootomo-style bf16x3 emulation of fp32 GEMM)
// ---------------------------------------------------------------------------
__device__ __forceinline__ void split8(v16bf& hi, v16bf& lo, int base, f4 a, f4 b)
{
    float v[8] = {a.x, a.y, a.z, a.w, b.x, b.y, b.z, b.w};
#pragma unroll
    for (int i = 0; i < 8; ++i) {
        __bf16 h = (__bf16)v[i];
        hi[base + i] = h;
        lo[base + i] = (__bf16)(v[i] - (float)h);
    }
}

// ---------------------------------------------------------------------------
// Kernel 2: MaxSim scoring. One wave32 per (b, candidate).
//   scores[b,j] = mean_q max_d  sum_h Q[b,q,h] * Doc[pid, d, h]
// A (16x32 bf16) = Q tile, B (32x16 bf16) = Doc^T tile, C = f32 16x16.
// ---------------------------------------------------------------------------
__global__ __launch_bounds__(256) void maxsim_score_kernel(
    const float* __restrict__ q, const float* __restrict__ docs,
    const int* __restrict__ pids, float* __restrict__ scores)
{
    const int lane = threadIdx.x & 31;
    const int wave = threadIdx.x >> 5;
    const int cand = blockIdx.x * 8 + wave;      // < BATCH*NTOPK
    const int b    = cand >> 10;                 // NTOPK == 1024
    const int pid  = pids[cand];                 // wave-uniform

    if (pid < 0) {                               // duplicate slot -> -inf
        if (lane == 0) scores[cand] = -__builtin_inff();
        return;
    }

    const int half = lane >> 4;                  // 0: lanes 0-15, 1: lanes 16-31
    const int row  = lane & 15;

    // ---- A fragments (Q): [mtile][kstep], hi+lo split -------------------
    // lane (half,row): element j holds K = kk*32 + 16*half + j
    v16bf a_hi[2][4], a_lo[2][4];
#pragma unroll
    for (int m = 0; m < 2; ++m) {
        const float* qr = q + (size_t)(b * QLEN + m * 16 + row) * DIM;
#pragma unroll
        for (int kk = 0; kk < 4; ++kk) {
            const float* p = qr + kk * 32 + 16 * half;
            f4 x0 = *(const f4*)(p);
            f4 x1 = *(const f4*)(p + 4);
            f4 x2 = *(const f4*)(p + 8);
            f4 x3 = *(const f4*)(p + 12);
            split8(a_hi[m][kk], a_lo[m][kk], 0, x0, x1);
            split8(a_hi[m][kk], a_lo[m][kk], 8, x2, x3);
        }
    }

    const float* dbase = docs + (size_t)pid * (DOC_LEN * DIM);

    float vmax[2][8];
#pragma unroll
    for (int m = 0; m < 2; ++m)
#pragma unroll
        for (int r = 0; r < 8; ++r) vmax[m][r] = -__builtin_inff();

    for (int n = 0; n < 8; ++n) {                // 8 N-tiles over DOC_LEN
        // B layout: lane (half,row): element j holds B[K=16*half+j][N=row]
        //   = Doc[token = n*16+row][dim = kk*32 + 16*half + j]   (contiguous!)
        const float* dr = dbase + (size_t)(n * 16 + row) * DIM + 16 * half;
        if (n < 7) __builtin_prefetch(dr + 16 * DIM, 0, 0);  // next tile row

        v16bf b_hi[4], b_lo[4];
#pragma unroll
        for (int kk = 0; kk < 4; ++kk) {
            const float* p = dr + kk * 32;
            f4 x0 = *(const f4*)(p);
            f4 x1 = *(const f4*)(p + 4);
            f4 x2 = *(const f4*)(p + 8);
            f4 x3 = *(const f4*)(p + 12);
            split8(b_hi[kk], b_lo[kk], 0, x0, x1);
            split8(b_hi[kk], b_lo[kk], 8, x2, x3);
        }

#pragma unroll
        for (int m = 0; m < 2; ++m) {
            v8f c = {};
#pragma unroll
            for (int kk = 0; kk < 4; ++kk) {
                // bf16x3: hi*lo + lo*hi (correction) then hi*hi
                c = __builtin_amdgcn_wmma_f32_16x16x32_bf16(
                        false, a_hi[m][kk], false, b_lo[kk], (short)0, c, false, false);
                c = __builtin_amdgcn_wmma_f32_16x16x32_bf16(
                        false, a_lo[m][kk], false, b_hi[kk], (short)0, c, false, false);
                c = __builtin_amdgcn_wmma_f32_16x16x32_bf16(
                        false, a_hi[m][kk], false, b_hi[kk], (short)0, c, false, false);
            }
            // C layout: lane holds N=row, element r -> M = m*16 + 8*half + r
#pragma unroll
            for (int r = 0; r < 8; ++r) vmax[m][r] = fmaxf(vmax[m][r], c[r]);
        }
    }

    // max over the 16 doc-token lanes of each half (all N values)
#pragma unroll
    for (int m = 0; m < 2; ++m)
#pragma unroll
        for (int r = 0; r < 8; ++r) {
            float v = vmax[m][r];
            v = fmaxf(v, __shfl_xor(v, 1, 32));
            v = fmaxf(v, __shfl_xor(v, 2, 32));
            v = fmaxf(v, __shfl_xor(v, 4, 32));
            v = fmaxf(v, __shfl_xor(v, 8, 32));
            vmax[m][r] = v;
        }

    // mean over q: lane sums its half's 16 q rows, then cross-half add
    float s = 0.0f;
#pragma unroll
    for (int m = 0; m < 2; ++m)
#pragma unroll
        for (int r = 0; r < 8; ++r) s += vmax[m][r];
    s += __shfl_xor(s, 16, 32);
    s *= (1.0f / 32.0f);

    if (lane == 0) scores[cand] = s;
}

// ---------------------------------------------------------------------------
// Kernel 3: top-10 per batch row (iterative argmax; lower index wins ties,
// matching jax.lax.top_k). Output: pids (as float) then scores.
// ---------------------------------------------------------------------------
__global__ __launch_bounds__(256) void topk_kernel(
    const int* __restrict__ pids, const float* __restrict__ scores,
    float* __restrict__ out)
{
    __shared__ float ls[NTOPK];
    __shared__ float rs[256];
    __shared__ int   ri[256];
    const int b = blockIdx.x;
    const int tid = threadIdx.x;

    for (int i = tid; i < NTOPK; i += 256) ls[i] = scores[b * NTOPK + i];
    __syncthreads();

    for (int t = 0; t < KOUT; ++t) {
        float best = -__builtin_inff();
        int bi = 0x7fffffff;
        for (int i = tid; i < NTOPK; i += 256) {
            float v = ls[i];
            if (v > best || (v == best && i < bi)) { best = v; bi = i; }
        }
        rs[tid] = best; ri[tid] = bi;
        __syncthreads();
        for (int off = 128; off > 0; off >>= 1) {
            if (tid < off) {
                float v = rs[tid + off]; int i2 = ri[tid + off];
                if (v > rs[tid] || (v == rs[tid] && i2 < ri[tid])) {
                    rs[tid] = v; ri[tid] = i2;
                }
            }
            __syncthreads();
        }
        if (tid == 0) {
            int w = ri[0];
            out[b * KOUT + t] = (float)pids[b * NTOPK + w];   // pid output
            out[BATCH * KOUT + b * KOUT + t] = rs[0];          // score output
            ls[w] = -__builtin_inff();                         // remove winner
        }
        __syncthreads();
    }
}

// ---------------------------------------------------------------------------
extern "C" void kernel_launch(void* const* d_in, const int* in_sizes, int n_in,
                              void* d_out, int out_size, void* d_ws, size_t ws_size,
                              hipStream_t stream)
{
    const float* q_vectors = (const float*)d_in[0];   // [B, QLEN, DIM]
    const float* vectors   = (const float*)d_in[1];   // [N_DOCS, DOC_LEN, DIM]
    const int*   emb2pid   = (const int*)d_in[2];     // [N_DOCS*DOC_LEN]
    const int*   topk_idx  = (const int*)d_in[3];     // [B, NTOPK]
    // d_in[4] = k (== 10, compile-time constant here)

    int*   pids_ws   = (int*)d_ws;                                    // 64 KB
    float* scores_ws = (float*)((char*)d_ws + BATCH * NTOPK * sizeof(int));

    sort_dedup_kernel<<<BATCH, 256, 0, stream>>>(topk_idx, emb2pid, pids_ws);

    // one wave32 per candidate; 8 waves per block
    maxsim_score_kernel<<<(BATCH * NTOPK) / 8, 256, 0, stream>>>(
        q_vectors, vectors, pids_ws, scores_ws);

    topk_kernel<<<BATCH, 256, 0, stream>>>(pids_ws, scores_ws, (float*)d_out);
}
